// Manifold_Net_55405078118629
// MI455X (gfx1250) — matrix operand order repaired and checked by reading, not compiled, not run on hardware
//
#include <hip/hip_runtime.h>

// ---------------------------------------------------------------------------
// Types for CDNA5 WMMA (wave32)
// ---------------------------------------------------------------------------
typedef __bf16 bf16;
typedef __attribute__((ext_vector_type(16))) __bf16 v16bf;
typedef __attribute__((ext_vector_type(8)))  __bf16 v8bf;
typedef __attribute__((ext_vector_type(8)))  float  v8f;

union VB16 { v16bf v; v8bf h[2]; bf16 e[16]; };

#define NB_ 1
#define NT_ 24
#define NX_ 256
#define NY_ 256
#define NITER_ 3
#define NF_ 16
#define LRELU_ 0.3f
#define NPTS_ ((long)NT_ * NX_ * NY_)   // 1,572,864

__device__ __forceinline__ unsigned pack_bf16x2(float a, float b)
{
    union { bf16 h[2]; unsigned u; } p;
    p.h[0] = (bf16)a; p.h[1] = (bf16)b;
    return p.u;
}

// ---------------------------------------------------------------------------
// Shared WMMA compute micro-kernel: one 64x64 block tile, fragments from LDS.
// 8 waves = 4(M) x 2(N); per wave: one 16-row strip x two 16-col subtiles,
// 4 f32 accumulators per subtile (rr,ii,ri,ir).
// ---------------------------------------------------------------------------
__device__ __forceinline__ void wmma_tile_step(
    const bf16 (*As_r)[40], const bf16 (*As_i)[40],
    const bf16 (*Bt_r)[40], const bf16 (*Bt_i)[40],
    int wm, int wn, int khalf, int mrow, int ncol,
    v8f rr[2], v8f ii[2], v8f ri[2], v8f ir[2])
{
    VB16 a_r, a_i;
    const int arow = wm * 16 + mrow;
    a_r.h[0] = *(const v8bf*)&As_r[arow][khalf * 8];
    a_r.h[1] = *(const v8bf*)&As_r[arow][16 + khalf * 8];
    a_i.h[0] = *(const v8bf*)&As_i[arow][khalf * 8];
    a_i.h[1] = *(const v8bf*)&As_i[arow][16 + khalf * 8];
#pragma unroll
    for (int s = 0; s < 2; ++s) {
        VB16 b_r, b_i;
        const int bcol = wn * 32 + s * 16 + ncol;
        b_r.h[0] = *(const v8bf*)&Bt_r[bcol][khalf * 8];
        b_r.h[1] = *(const v8bf*)&Bt_r[bcol][16 + khalf * 8];
        b_i.h[0] = *(const v8bf*)&Bt_i[bcol][khalf * 8];
        b_i.h[1] = *(const v8bf*)&Bt_i[bcol][16 + khalf * 8];
        rr[s] = __builtin_amdgcn_wmma_f32_16x16x32_bf16(false, a_r.v, false, b_r.v, (short)0, rr[s], false, false);
        ii[s] = __builtin_amdgcn_wmma_f32_16x16x32_bf16(false, a_i.v, false, b_i.v, (short)0, ii[s], false, false);
        ri[s] = __builtin_amdgcn_wmma_f32_16x16x32_bf16(false, a_r.v, false, b_i.v, (short)0, ri[s], false, false);
        ir[s] = __builtin_amdgcn_wmma_f32_16x16x32_bf16(false, a_i.v, false, b_r.v, (short)0, ir[s], false, false);
    }
}

// ---------------------------------------------------------------------------
// Generic complex GEMM, bf16 WMMA, f32 accumulate. C = op(A)*op(B).
// OPA: 0=N ; 1=conj-transpose.  OPB: 0=N ; 1=conj-transpose ; 2=conj.
// ACC: epilogue atomicAdd.  Requires even K,N,lda,ldb (true at all call sites).
// Staging is branch-free: clamped (always-legal) loads + value selects.
// ---------------------------------------------------------------------------
template<int OPA, int OPB, bool ACC>
__global__ __launch_bounds__(256)
void zgemm_wmma(const float* __restrict__ Ar, const float* __restrict__ Ai, int lda, long long sA,
                const float* __restrict__ Br, const float* __restrict__ Bi, int ldb, long long sB,
                float* __restrict__ Cr, float* __restrict__ Ci, int ldc, long long sC,
                int M, int N, int K)
{
    __shared__ alignas(16) bf16 As_r[64][40];   // [m][k]
    __shared__ alignas(16) bf16 As_i[64][40];
    __shared__ alignas(16) bf16 Bt_r[64][40];   // [n][k]  (transposed)
    __shared__ alignas(16) bf16 Bt_i[64][40];

    const int bz = blockIdx.z;
    Ar += (long long)bz * sA;  Ai += (long long)bz * sA;
    Br += (long long)bz * sB;  Bi += (long long)bz * sB;
    Cr += (long long)bz * sC;  Ci += (long long)bz * sC;

    const int tid   = threadIdx.x;
    const int wid   = tid >> 5;
    const int lane  = tid & 31;
    const int wm    = wid >> 1;
    const int wn    = wid & 1;
    const int khalf = lane >> 4;
    const int mrow  = lane & 15;
    const int ncol  = lane & 15;

    const int m0 = blockIdx.y * 64;
    const int n0 = blockIdx.x * 64;

    v8f rr[2], ii[2], ri[2], ir[2];
#pragma unroll
    for (int s = 0; s < 2; ++s) {
        rr[s] = (v8f){0.f,0.f,0.f,0.f,0.f,0.f,0.f,0.f};
        ii[s] = rr[s]; ri[s] = rr[s]; ir[s] = rr[s];
    }

    for (int k0 = 0; k0 < K; k0 += 32) {
        if (k0 + 32 < K && tid == 0) {
            if constexpr (OPA == 0) __builtin_prefetch(Ar + (long long)m0 * lda + k0 + 32, 0, 0);
            else                    __builtin_prefetch(Ar + (long long)(k0 + 32) * lda + m0, 0, 0);
            if constexpr (OPB == 1) __builtin_prefetch(Br + (long long)n0 * ldb + k0 + 32, 0, 0);
            else                    __builtin_prefetch(Br + (long long)(k0 + 32) * ldb + n0, 0, 0);
        }

        // ---- stage A tile (64m x 32k), branch-free ----
        if constexpr (OPA == 0) {
#pragma unroll
            for (int e = tid; e < 64 * 16; e += 256) {
                const int r = e >> 4, c = (e & 15) * 2;
                const int m = m0 + r, kk = k0 + c;
                const int mc = m < M ? m : M - 1;
                const int kc = kk < K ? kk : K - 2;
                const float2 fr = *(const float2*)(Ar + (long long)mc * lda + kc);
                const float2 fi = *(const float2*)(Ai + (long long)mc * lda + kc);
                const bool v = (m < M) && (kk < K);
                *(unsigned*)&As_r[r][c] = v ? pack_bf16x2(fr.x, fr.y) : 0u;
                *(unsigned*)&As_i[r][c] = v ? pack_bf16x2(fi.x, fi.y) : 0u;
            }
        } else { // OPA == 1: A(m,k) = conj(src[k*lda + m])
#pragma unroll
            for (int e = tid; e < 64 * 32; e += 256) {
                const int r = e >> 5, c = e & 31;
                const int m = m0 + r, kk = k0 + c;
                const int mc = m < M ? m : M - 1;
                const int kc = kk < K ? kk : K - 1;
                const long long idx = (long long)kc * lda + mc;
                const bool v = (m < M) && (kk < K);
                As_r[r][c] = (bf16)(v ?  Ar[idx] : 0.f);
                As_i[r][c] = (bf16)(v ? -Ai[idx] : 0.f);
            }
        }

        // ---- stage B tile into transposed LDS [n][k], branch-free ----
        if constexpr (OPB == 1) {
#pragma unroll
            for (int e = tid; e < 64 * 16; e += 256) {
                const int c = e >> 4, kp = (e & 15) * 2;
                const int n = n0 + c, kk = k0 + kp;
                const int nc = n < N ? n : N - 1;
                const int kc = kk < K ? kk : K - 2;
                const float2 fr = *(const float2*)(Br + (long long)nc * ldb + kc);
                const float2 fi = *(const float2*)(Bi + (long long)nc * ldb + kc);
                const bool v = (n < N) && (kk < K);
                *(unsigned*)&Bt_r[c][kp] = v ? pack_bf16x2( fr.x,  fr.y) : 0u;
                *(unsigned*)&Bt_i[c][kp] = v ? pack_bf16x2(-fi.x, -fi.y) : 0u;
            }
        } else {
#pragma unroll
            for (int e = tid; e < 32 * 32; e += 256) {
                const int kl = e >> 5, c = (e & 31) * 2;
                const int kk = k0 + kl, n = n0 + c;
                const int kc = kk < K ? kk : K - 1;
                const int nc = n < N ? n : N - 2;
                const float2 fr = *(const float2*)(Br + (long long)kc * ldb + nc);
                const float2 fi = *(const float2*)(Bi + (long long)kc * ldb + nc);
                const bool v = (kk < K) && (n < N);
                const float sgn = (OPB == 2) ? -1.f : 1.f;
                Bt_r[c][kl]     = (bf16)(v ? fr.x : 0.f);
                Bt_i[c][kl]     = (bf16)(v ? sgn * fi.x : 0.f);
                Bt_r[c + 1][kl] = (bf16)(v ? fr.y : 0.f);
                Bt_i[c + 1][kl] = (bf16)(v ? sgn * fi.y : 0.f);
            }
        }
        __syncthreads();
        wmma_tile_step(As_r, As_i, Bt_r, Bt_i, wm, wn, khalf, mrow, ncol, rr, ii, ri, ir);
        __syncthreads();
    }

#pragma unroll
    for (int s = 0; s < 2; ++s) {
        const int col = n0 + wn * 32 + s * 16 + ncol;
#pragma unroll
        for (int r = 0; r < 8; ++r) {
            const int row = m0 + wm * 16 + khalf * 8 + r;
            if (row < M && col < N) {
                const float cr = rr[s][r] - ii[s][r];
                const float ci = ri[s][r] + ir[s][r];
                const long long idx = (long long)row * ldc + col;
                if constexpr (ACC) { atomicAdd(&Cr[idx], cr); atomicAdd(&Ci[idx], ci); }
                else               { Cr[idx] = cr;            Ci[idx] = ci; }
            }
        }
    }
}

// ---------------------------------------------------------------------------
// FFT GEMM specialization: one operand is the constant 256x256 *symmetric*
// centered-DFT matrix, pre-stored in bf16. Symmetry => its transposed LDS
// tile is row-contiguous, so it is staged as a raw byte copy with CDNA5
// async global->LDS (ASYNCcnt), no VGPR round-trip / no convert.
// WMAT == 0: C = D * W   (along-y transform; W is B)
// WMAT == 1: C = W * D   (along-x transform; W is A)
// Dimensions are exact multiples of the tile (M%64==N%64==K%32==0).
// ---------------------------------------------------------------------------
template<int WMAT>
__global__ __launch_bounds__(256)
void zgemm_fft(const float* __restrict__ Dr, const float* __restrict__ Di, int ldd, long long sD,
               const bf16* __restrict__ Wr, const bf16* __restrict__ Wi,
               float* __restrict__ Cr, float* __restrict__ Ci, int ldc, long long sC,
               int M, int N, int K)
{
    __shared__ alignas(16) bf16 As_r[64][40];
    __shared__ alignas(16) bf16 As_i[64][40];
    __shared__ alignas(16) bf16 Bt_r[64][40];
    __shared__ alignas(16) bf16 Bt_i[64][40];

    const int bz = blockIdx.z;
    Dr += (long long)bz * sD;  Di += (long long)bz * sD;
    Cr += (long long)bz * sC;  Ci += (long long)bz * sC;

    const int tid   = threadIdx.x;
    const int wid   = tid >> 5;
    const int lane  = tid & 31;
    const int wm    = wid >> 1;
    const int wn    = wid & 1;
    const int khalf = lane >> 4;
    const int mrow  = lane & 15;
    const int ncol  = lane & 15;

    const int m0 = blockIdx.y * 64;
    const int n0 = blockIdx.x * 64;

    v8f rr[2], ii[2], ri[2], ir[2];
#pragma unroll
    for (int s = 0; s < 2; ++s) {
        rr[s] = (v8f){0.f,0.f,0.f,0.f,0.f,0.f,0.f,0.f};
        ii[s] = rr[s]; ri[s] = rr[s]; ir[s] = rr[s];
    }

    for (int k0 = 0; k0 < K; k0 += 32) {
        // ---- async-stage the W tile: 256 lanes x 16B = 4KB per component ----
        {
            const int row = tid >> 2, seg = tid & 3;            // 64 rows x 4 segments
            const int gro = (WMAT == 0 ? n0 : m0) + row;        // W row (symmetric!)
            const bf16* gr = Wr + (long long)gro * 256 + k0 + seg * 8;
            const bf16* gi = Wi + (long long)gro * 256 + k0 + seg * 8;
            bf16* lr = (WMAT == 0) ? &Bt_r[row][seg * 8] : &As_r[row][seg * 8];
            bf16* li = (WMAT == 0) ? &Bt_i[row][seg * 8] : &As_i[row][seg * 8];
            const unsigned lro = (unsigned)(size_t)lr;
            const unsigned lio = (unsigned)(size_t)li;
            asm volatile("global_load_async_to_lds_b128 %0, %1, off\n\t"
                         "global_load_async_to_lds_b128 %2, %3, off"
                         :: "v"(lro), "v"(gr), "v"(lio), "v"(gi)
                         : "memory");
        }
        // ---- stage the fp32 data tile (dims exact, no bounds needed) ----
        if constexpr (WMAT == 0) {          // data is A [m][k]
#pragma unroll
            for (int e = tid; e < 64 * 16; e += 256) {
                const int r = e >> 4, c = (e & 15) * 2;
                const float2 fr = *(const float2*)(Dr + (long long)(m0 + r) * ldd + k0 + c);
                const float2 fi = *(const float2*)(Di + (long long)(m0 + r) * ldd + k0 + c);
                *(unsigned*)&As_r[r][c] = pack_bf16x2(fr.x, fr.y);
                *(unsigned*)&As_i[r][c] = pack_bf16x2(fi.x, fi.y);
            }
        } else {                            // data is B: Bt[n][k] = D[k*ldd+n]
#pragma unroll
            for (int e = tid; e < 32 * 32; e += 256) {
                const int kl = e >> 5, c = (e & 31) * 2;
                const float2 fr = *(const float2*)(Dr + (long long)(k0 + kl) * ldd + n0 + c);
                const float2 fi = *(const float2*)(Di + (long long)(k0 + kl) * ldd + n0 + c);
                Bt_r[c][kl]     = (bf16)fr.x;  Bt_i[c][kl]     = (bf16)fi.x;
                Bt_r[c + 1][kl] = (bf16)fr.y;  Bt_i[c + 1][kl] = (bf16)fi.y;
            }
        }
        asm volatile("s_wait_asynccnt 0" ::: "memory");
        __syncthreads();
        wmma_tile_step(As_r, As_i, Bt_r, Bt_i, wm, wn, khalf, mrow, ncol, rr, ii, ri, ir);
        __syncthreads();
    }

#pragma unroll
    for (int s = 0; s < 2; ++s) {
        const int col = n0 + wn * 32 + s * 16 + ncol;
#pragma unroll
        for (int r = 0; r < 8; ++r) {
            const int row = m0 + wm * 16 + khalf * 8 + r;
            const long long idx = (long long)row * ldc + col;
            Cr[idx] = rr[s][r] - ii[s][r];
            Ci[idx] = ri[s][r] + ir[s][r];
        }
    }
}

// ---------------------------------------------------------------------------
// 3-D conv (3x3x3, SAME, 16->16 ch) as implicit GEMM on WMMA.
// Branch-free A-fragments: unconditional offset arithmetic + 64-bit select
// (v_cndmask) between the real offset and a zeroed scratch line.
// ---------------------------------------------------------------------------
__global__ __launch_bounds__(256)
void conv3d_wmma(const bf16* __restrict__ in, const bf16* __restrict__ w,
                 bf16* __restrict__ out, long long zrel, int lrelu)
{
    __shared__ alignas(16) bf16 wl[28 * 256];
    for (int e = threadIdx.x; e < 28 * 256; e += 256) wl[e] = w[e];
    __syncthreads();

    const int tid   = threadIdx.x;
    const int wid   = tid >> 5;
    const int lane  = tid & 31;
    const int khalf = lane >> 4;
    const int mrow  = lane & 15;
    const int ncol  = lane & 15;

    const int t  = blockIdx.z;
    const int x  = blockIdx.y;
    const int y0 = blockIdx.x * 128 + wid * 16;

    v8f acc = (v8f){0.f,0.f,0.f,0.f,0.f,0.f,0.f,0.f};

    for (int kk = 0; kk < 14; ++kk) {
        const int tapA = 2 * kk, tapB = 2 * kk + 1;
        VB16 a, b;
#pragma unroll
        for (int h = 0; h < 2; ++h) {
            const int tap = h ? tapB : tapA;
            const int dt = tap / 9 - 1, dx = (tap / 3) % 3 - 1, dy = tap % 3 - 1;
            const int tt = t + dt, xx = x + dx, yy = y0 + mrow + dy;
            const bool ok = (tap < 27) && ((unsigned)tt < (unsigned)NT_) &&
                            ((unsigned)xx < (unsigned)NX_) && ((unsigned)yy < (unsigned)NY_);
            const long long roff =
                (((long long)tt * NX_ + xx) * NY_ + yy) * NF_ + khalf * 8;
            const long long off = ok ? roff : zrel;       // v_cndmask, no branch
            a.h[h] = *(const v8bf*)(in + off);            // global_load_b128
        }
        {
            const bf16* wp = &wl[tapA * 256 + ncol * 16 + khalf * 8];  // [tap][co][ci]
            b.h[0] = *(const v8bf*)wp;                    // ds_load_b128
            b.h[1] = *(const v8bf*)(wp + 256);
        }
        acc = __builtin_amdgcn_wmma_f32_16x16x32_bf16(false, a.v, false, b.v, (short)0, acc, false, false);
    }

#pragma unroll
    for (int r = 0; r < 8; ++r) {
        const int y = y0 + khalf * 8 + r;
        float v = acc[r];
        if (lrelu) v = v > 0.f ? v : LRELU_ * v;
        out[(((long long)t * NX_ + x) * NY_ + y) * NF_ + ncol] = (bf16)v;
    }
}

// ---------------------------------------------------------------------------
// Small helper kernels
// ---------------------------------------------------------------------------
__device__ float block_reduce_sum(float v, float* sm)
{
    for (int off = 16; off > 0; off >>= 1) v += __shfl_down(v, off, 32);
    const int lane = threadIdx.x & 31, wid = threadIdx.x >> 5;
    if (lane == 0) sm[wid] = v;
    __syncthreads();
    v = (threadIdx.x < 8) ? sm[threadIdx.x] : 0.f;
    if (wid == 0) for (int off = 4; off > 0; off >>= 1) v += __shfl_down(v, off, 32);
    if (threadIdx.x == 0) sm[0] = v;
    __syncthreads();
    const float r = sm[0];
    __syncthreads();
    return r;
}

// Modified Gram-Schmidt QR (in place) on complex n x k (planar, ld=k).
__global__ __launch_bounds__(256)
void mgs_qr(float* Qr, float* Qi, int n, int k, int ld)
{
    __shared__ float sm[8];
    __shared__ float sm2[8];
    const int tid = threadIdx.x;
    for (int j = 0; j < k; ++j) {
        float loc = 0.f;
        for (int i = tid; i < n; i += 256) {
            const float a = Qr[i * ld + j], b = Qi[i * ld + j];
            loc += a * a + b * b;
        }
        const float nrm = block_reduce_sum(loc, sm);
        const float inv = rsqrtf(fmaxf(nrm, 1e-20f));
        for (int i = tid; i < n; i += 256) { Qr[i * ld + j] *= inv; Qi[i * ld + j] *= inv; }
        __syncthreads();
        for (int jj = j + 1; jj < k; ++jj) {
            float lr = 0.f, li = 0.f;
            for (int i = tid; i < n; i += 256) {
                const float ar = Qr[i * ld + j],  ai = Qi[i * ld + j];
                const float br = Qr[i * ld + jj], bi = Qi[i * ld + jj];
                lr += ar * br + ai * bi;
                li += ar * bi - ai * br;
            }
            const float dr = block_reduce_sum(lr, sm);
            const float di = block_reduce_sum(li, sm2);
            for (int i = tid; i < n; i += 256) {
                const float ar = Qr[i * ld + j], ai = Qi[i * ld + j];
                Qr[i * ld + jj] -= dr * ar - di * ai;
                Qi[i * ld + jj] -= dr * ai + di * ar;
            }
            __syncthreads();
        }
    }
}

__global__ void init_q(float* Qr, float* Qi, int total, unsigned seed)
{
    const int idx = blockIdx.x * 256 + threadIdx.x;
    if (idx >= total) return;
    unsigned h = (unsigned)idx * 2654435761u ^ seed;
    h ^= h >> 13; h *= 0x85ebca6bu; h ^= h >> 16;
    unsigned h2 = h * 0x27d4eb2du ^ 0x165667b1u; h2 ^= h2 >> 15;
    Qr[idx] = ((float)(h  & 0xffff) / 65536.f) - 0.5f;
    Qi[idx] = ((float)(h2 & 0xffff) / 65536.f) - 0.5f;
}

// Centered orthonormal DFT in bf16: Wf[k][n] = exp(-2pi i (k-128)(n-128)/256)/16
__global__ void build_dft(bf16* Wfr, bf16* Wfi, bf16* Wbr, bf16* Wbi)
{
    const int idx = blockIdx.x * 256 + threadIdx.x;
    if (idx >= 256 * 256) return;
    const int kq = idx >> 8, nq = idx & 255;
    const int prod = (kq - 128) * (nq - 128);
    const int pm = ((prod % 256) + 256) % 256;
    const float ang = -2.0f * 3.14159265358979323846f * (float)pm / 256.0f;
    const float c = __cosf(ang) * (1.0f / 16.0f);
    const float s = __sinf(ang) * (1.0f / 16.0f);
    Wfr[idx] = (bf16)c; Wfi[idx] = (bf16)s;
    Wbr[idx] = (bf16)c; Wbi[idx] = (bf16)(-s);
}

// Repack conv weights -> bf16 [layer][28 taps][16 co][16 ci], zero padded.
__global__ void repack_weights(const float* __restrict__ wi, const float* __restrict__ wh,
                               const float* __restrict__ wo, bf16* __restrict__ dst)
{
    const int per = 28 * 16 * 16;
    const int idx = blockIdx.x * 256 + threadIdx.x;
    if (idx >= NITER_ * 5 * per) return;
    const int l = idx / per, e = idx % per;
    const int tap = e >> 8, co = (e >> 4) & 15, ci = e & 15;
    const int iter = l / 5, li = l % 5;
    float v = 0.f;
    if (tap < 27) {
        if (li == 0)      { if (ci < 2) v = wi[((iter * 27 + tap) * 2 + ci) * 16 + co]; }
        else if (li <= 3) { v = wh[(((iter * 3 + (li - 1)) * 27 + tap) * 16 + ci) * 16 + co]; }
        else              { if (co < 2) v = wo[((iter * 27 + tap) * 16 + ci) * 2 + co]; }
    }
    dst[idx] = (bf16)v;
}

__global__ void zero_f32(float* p, long n)
{
    const long i = (long)blockIdx.x * 256 + threadIdx.x;
    if (i < n) p[i] = 0.f;
}

__global__ void pack_act(const float* __restrict__ xr, const float* __restrict__ xi,
                         bf16* __restrict__ act, long n)
{
    const long i = (long)blockIdx.x * 256 + threadIdx.x;
    if (i >= n) return;
    union { v8bf v; bf16 e[8]; } z0, z1;
#pragma unroll
    for (int c = 0; c < 8; ++c) { z0.e[c] = (bf16)0.f; z1.e[c] = (bf16)0.f; }
    z0.e[0] = (bf16)xr[i];  z0.e[1] = (bf16)xi[i];
    *(v8bf*)(act + i * NF_)     = z0.v;
    *(v8bf*)(act + i * NF_ + 8) = z1.v;
}

__global__ void unpack_act(const bf16* __restrict__ act, float* __restrict__ xr,
                           float* __restrict__ xi, long n)
{
    const long i = (long)blockIdx.x * 256 + threadIdx.x;
    if (i >= n) return;
    union { unsigned u; bf16 h[2]; } p;
    p.u = *(const unsigned*)(act + i * NF_);
    xr[i] = (float)p.h[0];
    xi[i] = (float)p.h[1];
}

__global__ void kresid_kernel(float* Kr, float* Ki, const float* __restrict__ m,
                              const float* __restrict__ dr, const float* __restrict__ di, long n)
{
    const long i = (long)blockIdx.x * 256 + threadIdx.x;
    if (i >= n) return;
    const float mm = m[i];
    Kr[i] = (Kr[i] * mm - dr[i]) * mm;
    Ki[i] = (Ki[i] * mm - di[i]) * mm;
}

__global__ void dc_kernel(float* Kr, float* Ki, const float* __restrict__ m,
                          const float* __restrict__ dr, const float* __restrict__ di, long n)
{
    const long i = (long)blockIdx.x * 256 + threadIdx.x;
    if (i >= n) return;
    const float mm = m[i];
    Kr[i] = (1.f - mm) * Kr[i] + mm * dr[i];
    Ki[i] = (1.f - mm) * Ki[i] + mm * di[i];
}

__global__ void axpy_sub(const float* __restrict__ ar, const float* __restrict__ ai,
                         const float* __restrict__ br, const float* __restrict__ bi,
                         float* __restrict__ outr, float* __restrict__ outi,
                         const float* __restrict__ eta, int it, long n)
{
    const long i = (long)blockIdx.x * 256 + threadIdx.x;
    if (i >= n) return;
    const float e = eta[it];
    outr[i] = ar[i] - e * br[i];
    outi[i] = ai[i] - e * bi[i];
}

__global__ void pack_out(const float* __restrict__ xr, const float* __restrict__ xi,
                         float* __restrict__ out, long n)
{
    const long i = (long)blockIdx.x * 256 + threadIdx.x;
    if (i >= n) return;
    out[2 * i]     = xr[i];
    out[2 * i + 1] = xi[i];
}

// ---------------------------------------------------------------------------
// Host-side orchestration
// ---------------------------------------------------------------------------
template<int OPA, int OPB, bool ACC>
static inline void zgemm(hipStream_t st, int M, int N, int K, int batch,
                         const float* Ar, const float* Ai, int lda, long long sA,
                         const float* Br, const float* Bi, int ldb, long long sB,
                         float* Cr, float* Ci, int ldc, long long sC)
{
    dim3 g((N + 63) / 64, (M + 63) / 64, batch);
    zgemm_wmma<OPA, OPB, ACC><<<g, dim3(256), 0, st>>>(Ar, Ai, lda, sA, Br, Bi, ldb, sB,
                                                       Cr, Ci, ldc, sC, M, N, K);
}

// fft2 (both axes) with bf16 symmetric DFT matrix W (async-staged in kernel).
static inline void fft2_mat(hipStream_t st,
                            const float* Xr, const float* Xi,
                            float* Tr, float* Ti, float* Or_, float* Oi,
                            const bf16* Wr, const bf16* Wi)
{
    const long long S = (long long)NX_ * NY_;
    // along y: rows = NT*NX, C = X * W
    dim3 g1(NY_ / 64, (NT_ * NX_) / 64, 1);
    zgemm_fft<0><<<g1, 256, 0, st>>>(Xr, Xi, NY_, 0, Wr, Wi, Tr, Ti, NY_, 0, NT_ * NX_, NY_, NY_);
    // along x: per t, C = W * X
    dim3 g2(NY_ / 64, NX_ / 64, NT_);
    zgemm_fft<1><<<g2, 256, 0, st>>>(Tr, Ti, NY_, S, Wr, Wi, Or_, Oi, NY_, S, NX_, NY_, NX_);
}

// top-k subspace projector P = Q Q^H from Gram G (n x n), k columns.
static inline void topk_proj(hipStream_t st, const float* Gr, const float* Gi, int n, int k,
                             float* Qr, float* Qi, float* Yr, float* Yi,
                             float* Pr, float* Pi, unsigned seed)
{
    const int tot = n * k;
    init_q<<<(tot + 255) / 256, 256, 0, st>>>(Qr, Qi, tot, seed);
    zgemm<0, 0, false>(st, n, k, n, 1, Gr, Gi, n, 0, Qr, Qi, k, 0, Yr, Yi, k, 0);
    mgs_qr<<<1, 256, 0, st>>>(Yr, Yi, n, k, k);
    zgemm<0, 0, false>(st, n, k, n, 1, Gr, Gi, n, 0, Yr, Yi, k, 0, Qr, Qi, k, 0);
    mgs_qr<<<1, 256, 0, st>>>(Qr, Qi, n, k, k);
    zgemm<0, 1, false>(st, n, n, k, 1, Qr, Qi, k, 0, Qr, Qi, k, 0, Pr, Pi, n, 0);
}

extern "C" void kernel_launch(void* const* d_in, const int* in_sizes, int n_in,
                              void* d_out, int out_size, void* d_ws, size_t ws_size,
                              hipStream_t stream)
{
    (void)in_sizes; (void)n_in; (void)out_size; (void)ws_size;
    const float* d_real = (const float*)d_in[0];
    const float* d_imag = (const float*)d_in[1];
    const float* mask   = (const float*)d_in[2];
    const float* w_in   = (const float*)d_in[3];
    const float* w_hid  = (const float*)d_in[4];
    const float* w_out  = (const float*)d_in[5];
    const float* eta    = (const float*)d_in[6];
    float* out = (float*)d_out;

    // ---- workspace arena ----
    char* w = (char*)d_ws;
    auto carve = [&](size_t bytes) -> void* {
        void* p = (void*)w;
        w += (bytes + 255) & ~(size_t)255;
        return p;
    };
    const size_t CB = (size_t)NPTS_ * sizeof(float);
    bf16* Wfr = (bf16*)carve(256 * 256 * 2);
    bf16* Wfi = (bf16*)carve(256 * 256 * 2);
    bf16* Wbr = (bf16*)carve(256 * 256 * 2);
    bf16* Wbi = (bf16*)carve(256 * 256 * 2);
    float* Xr  = (float*)carve(CB);  float* Xi  = (float*)carve(CB);
    float* XKr = (float*)carve(CB);  float* XKi = (float*)carve(CB);
    float* Kr  = (float*)carve(CB);  float* Ki  = (float*)carve(CB);
    float* Gr  = (float*)carve(CB);  float* Gi  = (float*)carve(CB);
    float* T1r = (float*)carve(CB);  float* T1i = (float*)carve(CB);
    float* T2r = (float*)carve(CB);  float* T2i = (float*)carve(CB);
    bf16* act_a = (bf16*)carve((size_t)NPTS_ * NF_ * sizeof(bf16));
    bf16* act_b = (bf16*)carve((size_t)NPTS_ * NF_ * sizeof(bf16));
    bf16* wpk   = (bf16*)carve((size_t)NITER_ * 5 * 28 * 256 * sizeof(bf16));
    bf16* zbuf  = (bf16*)carve(256);
    float* G1r = (float*)carve(256 * 256 * 4); float* G1i = (float*)carve(256 * 256 * 4);
    float* G2r = (float*)carve(256 * 256 * 4); float* G2i = (float*)carve(256 * 256 * 4);
    float* G3r = (float*)carve(24 * 24 * 4);   float* G3i = (float*)carve(24 * 24 * 4);
    float* Qbr = (float*)carve(256 * 128 * 4); float* Qbi = (float*)carve(256 * 128 * 4);
    float* Ybr = (float*)carve(256 * 128 * 4); float* Ybi = (float*)carve(256 * 128 * 4);
    float* P1r = (float*)carve(256 * 256 * 4); float* P1i = (float*)carve(256 * 256 * 4);
    float* P2r = (float*)carve(256 * 256 * 4); float* P2i = (float*)carve(256 * 256 * 4);
    float* P3r = (float*)carve(24 * 24 * 4);   float* P3i = (float*)carve(24 * 24 * 4);

    const long n = NPTS_;
    const int EB = (int)((n + 255) / 256);
    const long long S = (long long)NX_ * NY_;
    const dim3 cgrid(NY_ / 128, NX_, NT_);
    const long long zrel_a = (long long)(zbuf - act_a);
    const long long zrel_b = (long long)(zbuf - act_b);

    // ---- setup ----
    build_dft<<<(256 * 256 + 255) / 256, 256, 0, stream>>>(Wfr, Wfi, Wbr, Wbi);
    repack_weights<<<(NITER_ * 5 * 28 * 256 + 255) / 256, 256, 0, stream>>>(w_in, w_hid, w_out, wpk);
    zero_f32<<<1, 256, 0, stream>>>((float*)zbuf, 64);

    // x = ifft2c(d)
    fft2_mat(stream, d_real, d_imag, T1r, T1i, Xr, Xi, Wbr, Wbi);

    for (int it = 0; it < NITER_; ++it) {
        // ---- CNN: xk = cnn(x) ----
        pack_act<<<EB, 256, 0, stream>>>(Xr, Xi, act_a, n);
        const bf16* wl = wpk + (size_t)it * 5 * 28 * 256;
        conv3d_wmma<<<cgrid, 256, 0, stream>>>(act_a, wl + 0 * 28 * 256, act_b, zrel_a, 1);
        conv3d_wmma<<<cgrid, 256, 0, stream>>>(act_b, wl + 1 * 28 * 256, act_a, zrel_b, 1);
        conv3d_wmma<<<cgrid, 256, 0, stream>>>(act_a, wl + 2 * 28 * 256, act_b, zrel_a, 1);
        conv3d_wmma<<<cgrid, 256, 0, stream>>>(act_b, wl + 3 * 28 * 256, act_a, zrel_b, 1);
        conv3d_wmma<<<cgrid, 256, 0, stream>>>(act_a, wl + 4 * 28 * 256, act_b, zrel_a, 0);
        unpack_act<<<EB, 256, 0, stream>>>(act_b, XKr, XKi, n);

        // ---- g = ifft2c((fft2c(xk)*mask - d)*mask) ----
        fft2_mat(stream, XKr, XKi, T1r, T1i, Kr, Ki, Wfr, Wfi);
        kresid_kernel<<<EB, 256, 0, stream>>>(Kr, Ki, mask, d_real, d_imag, n);
        fft2_mat(stream, Kr, Ki, T1r, T1i, Gr, Gi, Wbr, Wbi);

        // ---- tangent == g ; retraction: x' = xk - eta*g, truncated Tucker ----
        axpy_sub<<<EB, 256, 0, stream>>>(XKr, XKi, Gr, Gi, T2r, T2i, eta, it, n);

        zero_f32<<<(256 * 256 + 255) / 256, 256, 0, stream>>>(G1r, 256 * 256);
        zero_f32<<<(256 * 256 + 255) / 256, 256, 0, stream>>>(G1i, 256 * 256);
        zero_f32<<<(256 * 256 + 255) / 256, 256, 0, stream>>>(G2r, 256 * 256);
        zero_f32<<<(256 * 256 + 255) / 256, 256, 0, stream>>>(G2i, 256 * 256);
        zgemm<0, 1, true>(stream, NX_, NX_, NY_, NT_, T2r, T2i, NY_, S, T2r, T2i, NY_, S, G1r, G1i, NX_, 0);
        zgemm<1, 0, true>(stream, NY_, NY_, NX_, NT_, T2r, T2i, NY_, S, T2r, T2i, NY_, S, G2r, G2i, NY_, 0);
        zgemm<0, 1, false>(stream, NT_, NT_, NX_ * NY_, 1, T2r, T2i, NX_ * NY_, 0, T2r, T2i, NX_ * NY_, 0, G3r, G3i, NT_, 0);

        topk_proj(stream, G1r, G1i, 256, 128, Qbr, Qbi, Ybr, Ybi, P1r, P1i, 0x1234u + it);
        topk_proj(stream, G2r, G2i, 256, 128, Qbr, Qbi, Ybr, Ybi, P2r, P2i, 0x5678u + it);
        topk_proj(stream, G3r, G3i, 24, 12,  Qbr, Qbi, Ybr, Ybi, P3r, P3i, 0x9abcu + it);

        zgemm<0, 0, false>(stream, NX_, NY_, NX_, NT_, P1r, P1i, NX_, 0, T2r, T2i, NY_, S, T1r, T1i, NY_, S);
        zgemm<0, 2, false>(stream, NT_ * NX_, NY_, NY_, 1, T1r, T1i, NY_, 0, P2r, P2i, NY_, 0, T2r, T2i, NY_, 0);
        zgemm<0, 0, false>(stream, NT_, NX_ * NY_, NT_, 1, P3r, P3i, NT_, 0, T2r, T2i, NX_ * NY_, 0, T1r, T1i, NX_ * NY_, 0);

        fft2_mat(stream, T1r, T1i, T2r, T2i, Kr, Ki, Wfr, Wfi);
        dc_kernel<<<EB, 256, 0, stream>>>(Kr, Ki, mask, d_real, d_imag, n);
        fft2_mat(stream, Kr, Ki, T1r, T1i, Xr, Xi, Wbr, Wbi);
    }

    pack_out<<<EB, 256, 0, stream>>>(Xr, Xi, out, n);
}
